// HierarchicalMoEFFN_17428977287678
// MI455X (gfx1250) — matrix operand order repaired and checked by reading, not compiled, not run on hardware
//
#include <hip/hip_runtime.h>

typedef __bf16 bf16;
typedef __attribute__((ext_vector_type(8)))  __bf16 v8bf;
typedef __attribute__((ext_vector_type(16))) __bf16 v16bf;
typedef __attribute__((ext_vector_type(8)))  float  v8f;

#define BATCH_ 8
#define T_     1024
#define NT_    8192      /* BATCH_*T_ */
#define D_     1024
#define DFF_   2048
#define NH_    4
#define HD_    256

__device__ __forceinline__ v16bf cat8(v8bf lo, v8bf hi) {
  v16bf r;
#pragma unroll
  for (int i = 0; i < 8; ++i) { r[i] = lo[i]; r[i + 8] = hi[i]; }
  return r;
}

// CDNA5 async memory->LDS copy (ASYNCcnt-tracked), 16 bytes per lane.
__device__ __forceinline__ void async_b128(bf16* lds_ptr, const bf16* gptr) {
  const unsigned l = (unsigned)(uintptr_t)lds_ptr;  // LDS aperture: low 32 bits
  asm volatile("global_load_async_to_lds_b128 %0, %1, off"
               :: "v"(l), "v"(gptr) : "memory");
}
__device__ __forceinline__ void wait_async0() {
  asm volatile("s_wait_asynccnt 0" ::: "memory");
}

// ---------------------------------------------------------------------------
// Generic bf16 GEMM:  C(M,N) = A(M,K) * Bw(N,K)^T  (+ bias[N])
// tile 128x128x32, 8 waves, wave tile 32x64 (2x4 wmma tiles).
// Double-buffered LDS fed by global_load_async_to_lds_b128; one barrier/iter.
// mode 0: store f32, mode 1: add f32, mode 2: store bf16
// ---------------------------------------------------------------------------
__global__ __launch_bounds__(256)
void gemm_bf16_kernel(const bf16* __restrict__ A, const bf16* __restrict__ Bw,
                      const float* __restrict__ bias, void* __restrict__ C,
                      int M, int N, int K, int mode) {
  __shared__ __align__(16) bf16 As[2][128 * 32];
  __shared__ __align__(16) bf16 Bs[2][128 * 32];
  const int tid  = threadIdx.x;
  const int lane = tid & 31;
  const int wid  = tid >> 5;
  const int wm   = wid & 3;       // 4 row groups of 32 rows
  const int wn   = wid >> 2;      // 2 col groups of 64 cols
  const int bm   = blockIdx.y * 128;
  const int bn   = blockIdx.x * 128;

  v8f acc[2][4];
#pragma unroll
  for (int a = 0; a < 2; ++a)
#pragma unroll
    for (int b = 0; b < 4; ++b)
#pragma unroll
      for (int i = 0; i < 8; ++i) acc[a][b][i] = 0.f;

  const int lrow  = tid >> 1;
  const int lcol  = (tid & 1) * 16;
  const int lm    = lane & 15;
  const int lhalf = (lane < 16) ? 0 : 8;   // unused; A-frag K also selected by koff
  const int koff  = (lane < 16) ? 0 : 16;  // B-frag K select

  const bf16* agbase = A  + (size_t)(bm + lrow) * K + lcol;
  const bf16* bgbase = Bw + (size_t)(bn + lrow) * K + lcol;
  const int ldsoff = lrow * 32 + lcol;

  // prologue: stream tile 0 into buffer 0
  async_b128(&As[0][ldsoff],     agbase);
  async_b128(&As[0][ldsoff + 8], agbase + 8);
  async_b128(&Bs[0][ldsoff],     bgbase);
  async_b128(&Bs[0][ldsoff + 8], bgbase + 8);
  wait_async0();
  __syncthreads();

  const int nk = K >> 5;
  for (int t = 0; t < nk; ++t) {
    const int cur = t & 1;
    const int nxt = cur ^ 1;
    if (t + 1 < nk) {  // prefetch next K-tile while computing this one
      const bf16* ag = agbase + (size_t)(t + 1) * 32;
      const bf16* bg = bgbase + (size_t)(t + 1) * 32;
      async_b128(&As[nxt][ldsoff],     ag);
      async_b128(&As[nxt][ldsoff + 8], ag + 8);
      async_b128(&Bs[nxt][ldsoff],     bg);
      async_b128(&Bs[nxt][ldsoff + 8], bg + 8);
    }

    v16bf afrag[2];
#pragma unroll
    for (int mt = 0; mt < 2; ++mt) {
      const int r = wm * 32 + mt * 16 + lm;
      afrag[mt] = cat8(*(const v8bf*)(&As[cur][r * 32 + koff]),
                       *(const v8bf*)(&As[cur][r * 32 + koff + 8]));
    }
    v16bf bfrag[4];
#pragma unroll
    for (int nt = 0; nt < 4; ++nt) {
      const int n = wn * 64 + nt * 16 + lm;
      bfrag[nt] = cat8(*(const v8bf*)(&Bs[cur][n * 32 + koff]),
                       *(const v8bf*)(&Bs[cur][n * 32 + koff + 8]));
    }
#pragma unroll
    for (int mt = 0; mt < 2; ++mt)
#pragma unroll
      for (int nt = 0; nt < 4; ++nt)
        acc[mt][nt] = __builtin_amdgcn_wmma_f32_16x16x32_bf16(
            false, afrag[mt], false, bfrag[nt], (short)0, acc[mt][nt], false, false);

    wait_async0();      // our async stores into nxt have landed in LDS
    __syncthreads();    // everyone done reading cur + writing nxt
  }

  const int rlo = (lane >> 4) * 8;
#pragma unroll
  for (int mt = 0; mt < 2; ++mt) {
#pragma unroll
    for (int nt = 0; nt < 4; ++nt) {
      const int gc = bn + wn * 64 + nt * 16 + lm;
      const float bv = bias ? bias[gc] : 0.f;
      const int gr0 = bm + wm * 32 + mt * 16 + rlo;
#pragma unroll
      for (int i = 0; i < 8; ++i) {
        const float val = acc[mt][nt][i] + bv;
        const size_t idx = (size_t)(gr0 + i) * N + gc;
        if (mode == 0)      ((float*)C)[idx] = val;
        else if (mode == 1) ((float*)C)[idx] += val;
        else                ((bf16*)C)[idx] = (bf16)val;
      }
    }
  }
}

// ---------------------------------------------------------------------------
// Causal flash attention, hd=256, bf16 in/out, WMMA for QK^T and P*V.
// grid: (T/128, B*H); 8 waves, each wave owns 16 query rows (no block barrier).
// Q,K: (B*T, H*hd) token-major; Vt: (B*H, hd, T) dim-major.
// ---------------------------------------------------------------------------
__global__ __launch_bounds__(256)
void attention_kernel(const bf16* __restrict__ Q, const bf16* __restrict__ Kb,
                      const bf16* __restrict__ Vt, bf16* __restrict__ O) {
  __shared__ __align__(16) bf16 plds[8][16 * 32];
  const int tid   = threadIdx.x;
  const int lane  = tid & 31;
  const int wid   = tid >> 5;
  const int bh    = blockIdx.y;
  const int b     = bh / NH_;
  const int h     = bh % NH_;
  const int qbase = blockIdx.x * 128 + wid * 16;
  const int Dm    = NH_ * HD_;
  const int lm    = lane & 15;
  const int lhalf = (lane < 16) ? 0 : 8;
  const int koff  = (lane < 16) ? 0 : 16;
  const int rbase = (lane >> 4) * 8;

  const bf16* qrow = Q + (size_t)(b * T_ + qbase + lm) * Dm + h * HD_;
  v16bf qfrag[8];
#pragma unroll
  for (int j = 0; j < 8; ++j) {
    const int d0 = j * 32 + koff;
    qfrag[j] = cat8(*(const v8bf*)(qrow + d0), *(const v8bf*)(qrow + d0 + 8));
  }

  v8f oacc[16];
#pragma unroll
  for (int n = 0; n < 16; ++n)
#pragma unroll
    for (int i = 0; i < 8; ++i) oacc[n][i] = 0.f;
  float mrow[8], lsum[8];
#pragma unroll
  for (int i = 0; i < 8; ++i) { mrow[i] = -1e30f; lsum[i] = 0.f; }

  const bf16* vb0 = Vt + (size_t)bh * HD_ * T_;

  for (int kb = 0; kb <= qbase + 15; kb += 32) {
    v8f s0, s1;
#pragma unroll
    for (int i = 0; i < 8; ++i) { s0[i] = 0.f; s1[i] = 0.f; }
    const bf16* k0 = Kb + (size_t)(b * T_ + kb + lm) * Dm + h * HD_ + koff;
    const bf16* k1 = k0 + (size_t)16 * Dm;
#pragma unroll
    for (int j = 0; j < 8; ++j) {
      v16bf kf0 = cat8(*(const v8bf*)(k0 + j * 32), *(const v8bf*)(k0 + j * 32 + 8));
      s0 = __builtin_amdgcn_wmma_f32_16x16x32_bf16(false, qfrag[j], false, kf0,
                                                   (short)0, s0, false, false);
      v16bf kf1 = cat8(*(const v8bf*)(k1 + j * 32), *(const v8bf*)(k1 + j * 32 + 8));
      s1 = __builtin_amdgcn_wmma_f32_16x16x32_bf16(false, qfrag[j], false, kf1,
                                                   (short)0, s1, false, false);
    }
    const int c0 = kb + lm, c1 = c0 + 16;
#pragma unroll
    for (int i = 0; i < 8; ++i) {
      const int qr = qbase + rbase + i;
      s0[i] = (c0 <= qr) ? s0[i] * 0.0625f : -1e30f;
      s1[i] = (c1 <= qr) ? s1[i] * 0.0625f : -1e30f;
    }
#pragma unroll
    for (int i = 0; i < 8; ++i) {
      float mx = fmaxf(s0[i], s1[i]);
#pragma unroll
      for (int off = 1; off < 16; off <<= 1) mx = fmaxf(mx, __shfl_xor(mx, off, 32));
      const float mn = fmaxf(mrow[i], mx);
      const float p0 = __expf(s0[i] - mn);
      const float p1 = __expf(s1[i] - mn);
      float rs = p0 + p1;
#pragma unroll
      for (int off = 1; off < 16; off <<= 1) rs += __shfl_xor(rs, off, 32);
      const float fac = __expf(mrow[i] - mn);
      lsum[i] = lsum[i] * fac + rs;
      mrow[i] = mn;
#pragma unroll
      for (int n = 0; n < 16; ++n) oacc[n][i] *= fac;
      plds[wid][(rbase + i) * 32 + lm]      = (bf16)p0;
      plds[wid][(rbase + i) * 32 + 16 + lm] = (bf16)p1;
    }
    v16bf pf = cat8(*(const v8bf*)(&plds[wid][lm * 32 + koff]),
                    *(const v8bf*)(&plds[wid][lm * 32 + koff + 8]));
#pragma unroll
    for (int n = 0; n < 16; ++n) {
      const bf16* vp = vb0 + (size_t)(n * 16 + lm) * T_ + kb + koff;
      v16bf vf = cat8(*(const v8bf*)vp, *(const v8bf*)(vp + 8));
      oacc[n] = __builtin_amdgcn_wmma_f32_16x16x32_bf16(false, pf, false, vf,
                                                        (short)0, oacc[n], false, false);
    }
  }

#pragma unroll
  for (int i = 0; i < 8; ++i) lsum[i] = 1.f / lsum[i];
#pragma unroll
  for (int n = 0; n < 16; ++n) {
    const int dim = h * HD_ + n * 16 + lm;
#pragma unroll
    for (int i = 0; i < 8; ++i) {
      const int qr = qbase + rbase + i;
      O[(size_t)(b * T_ + qr) * Dm + dim] = (bf16)(oacc[n][i] * lsum[i]);
    }
  }
}

// ---------------------------------------------------------------------------
// Per-row LayerNorm of f32 src -> bf16 dst; optional affine (s,b may be null).
// ---------------------------------------------------------------------------
__global__ __launch_bounds__(256)
void ln_to_bf16_kernel(const float* __restrict__ src, bf16* __restrict__ dst,
                       const float* __restrict__ s, const float* __restrict__ b) {
  const int row = blockIdx.x;
  const float* xr = src + (size_t)row * D_;
  __shared__ float red[256];
  float sum = 0.f, sq = 0.f;
  for (int d = threadIdx.x; d < D_; d += 256) { float v = xr[d]; sum += v; sq += v * v; }
  red[threadIdx.x] = sum; __syncthreads();
  for (int st = 128; st > 0; st >>= 1) {
    if (threadIdx.x < st) red[threadIdx.x] += red[threadIdx.x + st];
    __syncthreads();
  }
  const float mean = red[0] / (float)D_; __syncthreads();
  red[threadIdx.x] = sq; __syncthreads();
  for (int st = 128; st > 0; st >>= 1) {
    if (threadIdx.x < st) red[threadIdx.x] += red[threadIdx.x + st];
    __syncthreads();
  }
  const float var = red[0] / (float)D_ - mean * mean;
  const float rstd = rsqrtf(var + 1e-5f);
  for (int d = threadIdx.x; d < D_; d += 256) {
    float v = (xr[d] - mean) * rstd;
    if (s) v = v * s[d] + b[d];
    dst[(size_t)row * D_ + d] = (bf16)v;
  }
}

// ---------------------------------------------------------------------------
// Routing: per-token global (Q6 hexagram) + per-group router weights, and
// partial sums for the load-balance term.
// ---------------------------------------------------------------------------
__global__ __launch_bounds__(256)
void routing_kernel(const bf16* __restrict__ xhat,
                    const float* __restrict__ gr_s, const float* __restrict__ gr_b,
                    const float* __restrict__ q6w,  const float* __restrict__ gr_lt,
                    const float* __restrict__ grp_s, const float* __restrict__ grp_b,
                    const float* __restrict__ grp_w, const float* __restrict__ grp_bias,
                    float* __restrict__ gw, float* __restrict__ sparse,
                    float* __restrict__ sums) {
  const int t = blockIdx.x;
  const bf16* xr = xhat + (size_t)t * D_;
  const int lane = threadIdx.x & 31, wid = threadIdx.x >> 5;
  float p[12];
#pragma unroll
  for (int j = 0; j < 12; ++j) p[j] = 0.f;
  for (int d = threadIdx.x; d < D_; d += 256) {
    const float xh = (float)xr[d];
    const float hg = xh * gr_s[d] + gr_b[d];
#pragma unroll
    for (int j = 0; j < 6; ++j) p[j] += hg * q6w[j * D_ + d];
#pragma unroll
    for (int g = 0; g < 3; ++g) {
      const float hgr = xh * grp_s[g * D_ + d] + grp_b[g * D_ + d];
      p[6 + g * 2 + 0] += hgr * grp_w[(g * 2 + 0) * D_ + d];
      p[6 + g * 2 + 1] += hgr * grp_w[(g * 2 + 1) * D_ + d];
    }
  }
#pragma unroll
  for (int j = 0; j < 12; ++j)
#pragma unroll
    for (int off = 1; off < 32; off <<= 1) p[j] += __shfl_xor(p[j], off, 32);
  __shared__ float part[8][12];
  if (lane == 0)
    for (int j = 0; j < 12; ++j) part[wid][j] = p[j];
  __syncthreads();
  if (threadIdx.x == 0) {
    float d12[12];
    for (int j = 0; j < 12; ++j) {
      float a = 0.f;
      for (int w = 0; w < 8; ++w) a += part[w][j];
      d12[j] = a;
    }
    float sb[6];
    for (int j = 0; j < 6; ++j) sb[j] = tanhf(d12[j]);
    float temp = expf(gr_lt[0]);
    temp = fminf(fmaxf(temp, 0.1f), 5.0f);
    float lg[64], mx = -1e30f;
    for (int i = 0; i < 64; ++i) {
      float a = 0.f;
      for (int j = 0; j < 6; ++j) a += (((i >> (5 - j)) & 1) ? sb[j] : -sb[j]);
      lg[i] = a / temp;
      mx = fmaxf(mx, lg[i]);
    }
    float se = 0.f;
    for (int i = 0; i < 64; ++i) { lg[i] = expf(lg[i] - mx); se += lg[i]; }
    float sh[6] = {0.f, 0.f, 0.f, 0.f, 0.f, 0.f};
    for (int i = 0; i < 64; ++i) {
      const float w = lg[i] / se;
      for (int j = 0; j < 6; ++j) sh[j] += w * (((i >> (5 - j)) & 1) ? 1.f : -1.f);
    }
    const float c5 = 0.44721359549995794f;
    const float a0[6] = { c5,  c5,  c5,  c5,  c5, 0.f};
    const float a1[6] = {-0.5f, 0.5f, -0.5f, 0.f, 0.f, 0.5f};
    const float a2[6] = {-c5, -c5, 0.f, -c5, -c5, -c5};
    float gl[3] = {0.f, 0.f, 0.f};
    for (int j = 0; j < 6; ++j) {
      gl[0] += sh[j] * a0[j]; gl[1] += sh[j] * a1[j]; gl[2] += sh[j] * a2[j];
    }
    const float gm = fmaxf(gl[0], fmaxf(gl[1], gl[2]));
    float ge[3], gs = 0.f;
    for (int g = 0; g < 3; ++g) { ge[g] = expf(gl[g] - gm); gs += ge[g]; }
    for (int g = 0; g < 3; ++g) {
      const float v = ge[g] / gs;
      gw[t * 3 + g] = v;
      atomicAdd(&sums[g], v);
    }
    for (int g = 0; g < 3; ++g) {
      const float l0 = d12[6 + g * 2] + grp_bias[g * 2];
      const float l1 = d12[7 + g * 2] + grp_bias[g * 2 + 1];
      const float m2 = fmaxf(l0, l1);
      const float e0 = expf(l0 - m2), e1 = expf(l1 - m2);
      const float w0 = e0 / (e0 + e1), w1 = e1 / (e0 + e1);
      atomicAdd(&sums[3 + g * 2], w0);
      atomicAdd(&sums[4 + g * 2], w1);
      const float rn = 1.f / (w0 + w1 + 1e-8f);
      sparse[t * 6 + g * 2]     = w0 * rn;
      sparse[t * 6 + g * 2 + 1] = w1 * rn;
    }
  }
}

__global__ void finalize_lb_kernel(const float* __restrict__ sums, float* __restrict__ out_lb) {
  if (threadIdx.x == 0 && blockIdx.x == 0) {
    const float inv = 1.f / (float)NT_;
    float lb = 0.f;
    for (int i = 0; i < 9; ++i) {
      const float m = sums[i] * inv;
      lb += m * logf(m + 1e-8f);
    }
    const float uniform = 0.5f;
    const float threshold = uniform + fminf(0.15f, (1.f - uniform) * 0.3f);
    lb += 3.f * 0.1f * fmaxf(uniform - threshold, 0.f);  // == 0 in eval mode
    *out_lb = lb;
  }
}

// ---------------------------------------------------------------------------
// Small utility kernels
// ---------------------------------------------------------------------------
__global__ __launch_bounds__(256)
void fold_w_kernel(const float* __restrict__ src, const float* __restrict__ scale,
                   bf16* __restrict__ dst, size_t n, int K) {
  const size_t i = (size_t)blockIdx.x * 256 + threadIdx.x;
  if (i >= n) return;
  float v = src[i];
  if (scale) v *= scale[i % (size_t)K];
  dst[i] = (bf16)v;
}

__global__ __launch_bounds__(256)
void fold_bias_kernel(const float* __restrict__ W, const float* __restrict__ lnb,
                      float* __restrict__ dst, int K) {
  const int n = blockIdx.x;
  __shared__ float red[256];
  float s = 0.f;
  for (int k = threadIdx.x; k < K; k += 256) s += lnb[k] * W[(size_t)n * K + k];
  red[threadIdx.x] = s; __syncthreads();
  for (int st = 128; st > 0; st >>= 1) {
    if (threadIdx.x < st) red[threadIdx.x] += red[threadIdx.x + st];
    __syncthreads();
  }
  if (threadIdx.x == 0) dst[n] = red[0];
}

__global__ __launch_bounds__(256)
void swiglu_kernel(const float* __restrict__ g, const float* __restrict__ v,
                   bf16* __restrict__ h, size_t n) {
  const size_t i = (size_t)blockIdx.x * 256 + threadIdx.x;
  if (i >= n) return;
  const float gg = g[i];
  h[i] = (bf16)((gg / (1.f + __expf(-gg))) * v[i]);
}

// dst (+)= w[t*stride+idx] * src   over (NT, D)
__global__ __launch_bounds__(256)
void rowscale_accum_kernel(float* __restrict__ dst, const float* __restrict__ src,
                           const float* __restrict__ w, int stride, int idx, int add) {
  const size_t i = (size_t)blockIdx.x * 256 + threadIdx.x;
  if (i >= (size_t)NT_ * D_) return;
  const size_t t = i / D_;
  const float v = w[t * stride + idx] * src[i];
  if (add) dst[i] += v; else dst[i] = v;
}

__global__ __launch_bounds__(256)
void transpose_v_kernel(const bf16* __restrict__ v, bf16* __restrict__ vt) {
  const size_t i = (size_t)blockIdx.x * 256 + threadIdx.x;
  if (i >= (size_t)NT_ * D_) return;
  const int d = (int)(i % D_);
  const size_t tok = i / D_;
  const int b = (int)(tok / T_), tt = (int)(tok % T_);
  const int h = d / HD_, dd = d % HD_;
  vt[((size_t)(b * NH_ + h) * HD_ + dd) * T_ + tt] = v[i];
}

__global__ __launch_bounds__(256)
void crossed_kernel(const float* __restrict__ xa, const float* __restrict__ fwd,
                    const float* __restrict__ xb, const float* __restrict__ bwd,
                    const float* __restrict__ log_alpha, float* __restrict__ out) {
  const size_t i = (size_t)blockIdx.x * 256 + threadIdx.x;
  if (i >= (size_t)NT_ * D_) return;
  const float a = 1.f / (1.f + __expf(-log_alpha[0]));
  out[i] = a * (xa[i] + fwd[i]) + (1.f - a) * (xb[i] + bwd[i]);
}

__global__ __launch_bounds__(256)
void final_kernel(float* __restrict__ out, const float* __restrict__ crossed,
                  const float* __restrict__ y, const float* __restrict__ gate) {
  const size_t i = (size_t)blockIdx.x * 256 + threadIdx.x;
  if (i >= (size_t)NT_ * D_) return;
  const float gv = 1.f / (1.f + __expf(-gate[0]));
  out[i] += 0.5f * gv * (crossed[i] + y[i]);
}

// ---------------------------------------------------------------------------
// Host orchestration
// ---------------------------------------------------------------------------
extern "C" void kernel_launch(void* const* d_in, const int* in_sizes, int n_in,
                              void* d_out, int out_size, void* d_ws, size_t ws_size,
                              hipStream_t stream) {
  (void)in_sizes; (void)n_in; (void)out_size; (void)ws_size;
  const float* x           = (const float*)d_in[0];
  const float* gr_ln_s     = (const float*)d_in[1];
  const float* gr_ln_b     = (const float*)d_in[2];
  const float* gr_q6_w     = (const float*)d_in[3];
  const float* gr_log_temp = (const float*)d_in[4];
  const float* grp_ln_s    = (const float*)d_in[5];
  const float* grp_ln_b    = (const float*)d_in[6];
  const float* grp_w       = (const float*)d_in[7];
  const float* grp_b       = (const float*)d_in[8];
  const float* exp_ln_s    = (const float*)d_in[9];
  const float* exp_ln_b    = (const float*)d_in[10];
  const float* exp_gate    = (const float*)d_in[11];
  const float* exp_val     = (const float*)d_in[12];
  const float* exp_out     = (const float*)d_in[13];
  const float* br_qkv_f    = (const float*)d_in[14];
  const float* br_o_f      = (const float*)d_in[15];
  const float* br_qkv_b    = (const float*)d_in[16];
  const float* br_o_b      = (const float*)d_in[17];
  const float* br_lna_s    = (const float*)d_in[18];
  const float* br_lna_b    = (const float*)d_in[19];
  const float* br_lnb_s    = (const float*)d_in[20];
  const float* br_lnb_b    = (const float*)d_in[21];
  const float* br_lnf_s    = (const float*)d_in[22];
  const float* br_lnf_b    = (const float*)d_in[23];
  const float* br_ffg      = (const float*)d_in[24];
  const float* br_ffv      = (const float*)d_in[25];
  const float* br_ffo      = (const float*)d_in[26];
  const float* br_log_alpha= (const float*)d_in[27];
  const float* br_gate     = (const float*)d_in[28];

  float* out = (float*)d_out;

  char* ws = (char*)d_ws;
  size_t off = 0;
  auto alloc = [&](size_t bytes) -> char* {
    char* p = ws + off;
    off += (bytes + 255) & ~(size_t)255;
    return p;
  };
  const size_t NTD  = (size_t)NT_ * D_;
  const size_t NTF  = (size_t)NT_ * DFF_;

  bf16*  xhat    = (bf16*)alloc(NTD * 2);
  float* gwbuf   = (float*)alloc((size_t)NT_ * 3 * 4);
  float* sparse  = (float*)alloc((size_t)NT_ * 6 * 4);
  float* sums    = (float*)alloc(64);
  float* Gbuf    = (float*)alloc(NTF * 4);
  float* Vbuf    = (float*)alloc(NTF * 4);
  bf16*  Hbuf    = (bf16*)alloc(NTF * 2);
  float* Ybuf    = (float*)alloc(NTD * 4);
  float* go0     = (float*)alloc(NTD * 4);
  float* go1     = (float*)alloc(NTD * 4);
  float* go2     = (float*)alloc(NTD * 4);
  bf16*  na      = (bf16*)alloc(NTD * 2);
  bf16*  nb      = (bf16*)alloc(NTD * 2);
  bf16*  qb      = (bf16*)alloc(NTD * 2);
  bf16*  kb      = (bf16*)alloc(NTD * 2);
  bf16*  vb      = (bf16*)alloc(NTD * 2);
  bf16*  vt      = (bf16*)alloc(NTD * 2);
  float* crossed = (float*)alloc(NTD * 4);
  bf16*  wstage  = (bf16*)alloc((size_t)3 * D_ * D_ * 2);
  float* bstage  = (float*)alloc((size_t)DFF_ * 4);
  float* go[3]   = {go0, go1, go2};
  bf16*  att     = xhat;   // xhat dead after routing + experts
  bf16*  hf      = na;     // na dead after bwd qkv GEMMs
  float* fwdbuf  = Gbuf;   // G/V scratch free between expert and bridge-FFN phases
  float* bwdbuf  = Vbuf;

  auto gemm = [&](const bf16* A, const bf16* Bw, const float* bias, void* C,
                  int M, int N, int K, int mode) {
    gemm_bf16_kernel<<<dim3(N / 128, M / 128), 256, 0, stream>>>(A, Bw, bias, C, M, N, K, mode);
  };
  auto ewgrid = [](size_t n) { return (unsigned)((n + 255) / 256); };

  // ---- shared normalized activations + routing + lb ----
  hipMemsetAsync(sums, 0, 64, stream);
  ln_to_bf16_kernel<<<NT_, 256, 0, stream>>>(x, xhat, nullptr, nullptr);
  routing_kernel<<<NT_, 256, 0, stream>>>(xhat, gr_ln_s, gr_ln_b, gr_q6_w, gr_log_temp,
                                          grp_ln_s, grp_ln_b, grp_w, grp_b,
                                          gwbuf, sparse, sums);
  finalize_lb_kernel<<<1, 32, 0, stream>>>(sums, out + NTD);

  // ---- micro-experts (LN folded into weights; bias = ln_b @ W^T) ----
  for (int g = 0; g < 3; ++g) {
    for (int e = 0; e < 2; ++e) {
      const int ge = g * 2 + e;
      const float* lns = exp_ln_s + (size_t)ge * D_;
      const float* lnb = exp_ln_b + (size_t)ge * D_;
      const float* wg  = exp_gate + (size_t)ge * DFF_ * D_;
      const float* wv  = exp_val  + (size_t)ge * DFF_ * D_;
      const float* wo  = exp_out  + (size_t)ge * D_ * DFF_;
      fold_w_kernel<<<ewgrid((size_t)DFF_ * D_), 256, 0, stream>>>(wg, lns, wstage, (size_t)DFF_ * D_, D_);
      fold_bias_kernel<<<DFF_, 256, 0, stream>>>(wg, lnb, bstage, D_);
      gemm(xhat, wstage, bstage, Gbuf, NT_, DFF_, D_, 0);
      fold_w_kernel<<<ewgrid((size_t)DFF_ * D_), 256, 0, stream>>>(wv, lns, wstage, (size_t)DFF_ * D_, D_);
      fold_bias_kernel<<<DFF_, 256, 0, stream>>>(wv, lnb, bstage, D_);
      gemm(xhat, wstage, bstage, Vbuf, NT_, DFF_, D_, 0);
      swiglu_kernel<<<ewgrid(NTF), 256, 0, stream>>>(Gbuf, Vbuf, Hbuf, NTF);
      fold_w_kernel<<<ewgrid((size_t)D_ * DFF_), 256, 0, stream>>>(wo, nullptr, wstage, (size_t)D_ * DFF_, DFF_);
      gemm(Hbuf, wstage, nullptr, Ybuf, NT_, D_, DFF_, 0);
      rowscale_accum_kernel<<<ewgrid(NTD), 256, 0, stream>>>(go[g], Ybuf, sparse, 6, ge, e == 1);
    }
    rowscale_accum_kernel<<<ewgrid(NTD), 256, 0, stream>>>(out, go[g], gwbuf, 3, g, g > 0);
  }

  // ---- bridge: forward cross-attention (q from na; k,v from nb) ----
  ln_to_bf16_kernel<<<NT_, 256, 0, stream>>>(go0, na, br_lna_s, br_lna_b);
  ln_to_bf16_kernel<<<NT_, 256, 0, stream>>>(go2, nb, br_lnb_s, br_lnb_b);

  fold_w_kernel<<<ewgrid((size_t)3 * D_ * D_), 256, 0, stream>>>(br_qkv_f, nullptr, wstage, (size_t)3 * D_ * D_, D_);
  gemm(na, wstage,                       nullptr, qb, NT_, D_, D_, 2);
  gemm(nb, wstage + (size_t)D_ * D_,     nullptr, kb, NT_, D_, D_, 2);
  gemm(nb, wstage + (size_t)2 * D_ * D_, nullptr, vb, NT_, D_, D_, 2);
  transpose_v_kernel<<<ewgrid(NTD), 256, 0, stream>>>(vb, vt);
  attention_kernel<<<dim3(T_ / 128, BATCH_ * NH_), 256, 0, stream>>>(qb, kb, vt, att);
  fold_w_kernel<<<ewgrid((size_t)D_ * D_), 256, 0, stream>>>(br_o_f, nullptr, wstage, (size_t)D_ * D_, D_);
  gemm(att, wstage, nullptr, fwdbuf, NT_, D_, D_, 0);

  // ---- bridge: backward cross-attention (q from nb; k,v from na) ----
  fold_w_kernel<<<ewgrid((size_t)3 * D_ * D_), 256, 0, stream>>>(br_qkv_b, nullptr, wstage, (size_t)3 * D_ * D_, D_);
  gemm(nb, wstage,                       nullptr, qb, NT_, D_, D_, 2);
  gemm(na, wstage + (size_t)D_ * D_,     nullptr, kb, NT_, D_, D_, 2);
  gemm(na, wstage + (size_t)2 * D_ * D_, nullptr, vb, NT_, D_, D_, 2);
  transpose_v_kernel<<<ewgrid(NTD), 256, 0, stream>>>(vb, vt);
  attention_kernel<<<dim3(T_ / 128, BATCH_ * NH_), 256, 0, stream>>>(qb, kb, vt, att);
  fold_w_kernel<<<ewgrid((size_t)D_ * D_), 256, 0, stream>>>(br_o_b, nullptr, wstage, (size_t)D_ * D_, D_);
  gemm(att, wstage, nullptr, bwdbuf, NT_, D_, D_, 0);

  // ---- blend + bridge FFN + final ----
  crossed_kernel<<<ewgrid(NTD), 256, 0, stream>>>(go0, fwdbuf, go2, bwdbuf, br_log_alpha, crossed);
  ln_to_bf16_kernel<<<NT_, 256, 0, stream>>>(crossed, hf, br_lnf_s, br_lnf_b);
  fold_w_kernel<<<ewgrid((size_t)DFF_ * D_), 256, 0, stream>>>(br_ffg, nullptr, wstage, (size_t)DFF_ * D_, D_);
  gemm(hf, wstage, nullptr, Gbuf, NT_, DFF_, D_, 0);
  fold_w_kernel<<<ewgrid((size_t)DFF_ * D_), 256, 0, stream>>>(br_ffv, nullptr, wstage, (size_t)DFF_ * D_, D_);
  gemm(hf, wstage, nullptr, Vbuf, NT_, DFF_, D_, 0);
  swiglu_kernel<<<ewgrid(NTF), 256, 0, stream>>>(Gbuf, Vbuf, Hbuf, NTF);
  fold_w_kernel<<<ewgrid((size_t)D_ * DFF_), 256, 0, stream>>>(br_ffo, nullptr, wstage, (size_t)D_ * DFF_, DFF_);
  gemm(Hbuf, wstage, nullptr, Ybuf, NT_, D_, DFF_, 0);
  final_kernel<<<ewgrid(NTD), 256, 0, stream>>>(out, crossed, Ybuf, br_gate);
}
